// PyramidModule_24068996726847
// MI455X (gfx1250) — matrix operand order repaired and checked by reading, not compiled, not run on hardware
//
#include <hip/hip_runtime.h>
#include <hip/hip_bf16.h>

typedef _Float16 v16h __attribute__((ext_vector_type(16)));
typedef float    v8f  __attribute__((ext_vector_type(8)));

#define WMMA_F16(a, b, c) \
  __builtin_amdgcn_wmma_f32_16x16x32_f16(false, (a), false, (b), (short)0, (c), false, false)

// ---------------------------------------------------------------------------
// CDNA5 WMMA 16x16x32 f16 operand layouts (CDNA5 ISA):
//  A (16x32, MxK): lane L holds row M=L&15; halves h -> K = ((L>>4)<<4)+h (k order used here, not the ISA's table order).
//  B (32x16, KxN): lane L holds col N=L&15; halves h -> K = ((L>>4)<<4)+h (same k order as A, chosen here).
//  C/D (16x16 f32): elem v -> row = v + ((L>>4)<<3), col = L&15.
// ---------------------------------------------------------------------------
__device__ __forceinline__ int a_k(int lane, int h) {
  return ((lane >> 4) << 4) + h;
}

// Generic D = A(rows of 128-stride f16) x W^T (W row-major [N][K]) tile.
__device__ __forceinline__ v8f wmma_smem_w(const _Float16* As, int row0,
                                           const _Float16* W, int col0, int K) {
  int lane = threadIdx.x & 31;
  int am  = (row0 + (lane & 15)) * 128;
  int wn  = (col0 + (lane & 15)) * K;
  int bk0 = (lane >> 4) << 4;
  // Pull the weight panel line through WGP$ ahead of the per-half gathers
  // (lowers to global_prefetch_b8 on gfx1250).
  __builtin_prefetch(&W[wn], 0, 3);
  v8f c = {};
  for (int ks = 0; ks < K; ks += 32) {
    v16h a, b;
#pragma unroll
    for (int h = 0; h < 16; ++h) {
      a[h] = As[am + ks + a_k(lane, h)];
      b[h] = W[wn + ks + bk0 + h];
    }
    c = WMMA_F16(a, b, c);
  }
  return c;
}

// ---------------------------------------------------------------------------
// Small helpers
// ---------------------------------------------------------------------------
__global__ void zero_kernel(float* p, int n) {
  int i = blockIdx.x * blockDim.x + threadIdx.x;
  if (i < n) p[i] = 0.f;
}

// f32 [rows][srcK] -> f16 [rows][dstK], zero-padded along K.
__global__ void cvt_kernel(const float* src, _Float16* dst, int rows, int srcK, int dstK) {
  int i = blockIdx.x * blockDim.x + threadIdx.x;
  if (i >= rows * dstK) return;
  int r = i / dstK, k = i - r * dstK;
  dst[i] = (_Float16)((k < srcK) ? src[r * srcK + k] : 0.f);
}

__global__ void bn_final_kernel(const float* sum, const float* sumsq,
                                const float* g, const float* b, float* ss, float invP) {
  int c = threadIdx.x;  // 128
  float mu  = sum[c] * invP;
  float var = sumsq[c] * invP - mu * mu;
  float sc  = g[c] * rsqrtf(var + 1e-5f);
  ss[c] = sc;
  ss[128 + c] = b[c] - mu * sc;
}

// ---------------------------------------------------------------------------
// Ball grouping: one thread per query; scan the (batch-restricted) point
// range in ascending index order => exactly the nsample smallest valid ids.
// ---------------------------------------------------------------------------
__global__ void group_kernel(const float* __restrict__ xyz,
                             const float* __restrict__ q_xyz,
                             const float* __restrict__ q_r,
                             const int* __restrict__ xyz_cnt,
                             const int* __restrict__ q_cnt,
                             float* gx, int* gidx, float* d_empty,
                             int M, int ns, int NP) {
  int m = blockIdx.x * blockDim.x + threadIdx.x;
  if (m >= M) return;
  int qb    = (m < q_cnt[0]) ? 0 : 1;
  int start = (qb == 0) ? 0 : xyz_cnt[0];
  int end   = start + xyz_cnt[qb];
  if (end > NP) end = NP;
  float qx = q_xyz[m * 3 + 0], qy = q_xyz[m * 3 + 1], qz = q_xyz[m * 3 + 2];
  float r  = q_r[m];
  float r2 = r * r;
  int cnt = 0, first = -1;
  for (int j = start; j < end && cnt < ns; ++j) {
    float dx = xyz[j * 3 + 0] - qx;
    float dy = xyz[j * 3 + 1] - qy;
    float dz = xyz[j * 3 + 2] - qz;
    if (dx * dx + dy * dy + dz * dz < r2) {
      if (first < 0) first = j;
      int p = m * ns + cnt;
      gidx[p] = j;
      gx[p * 4 + 0] = dx; gx[p * 4 + 1] = dy; gx[p * 4 + 2] = dz; gx[p * 4 + 3] = 0.f;
      ++cnt;
    }
  }
  if (cnt == 0) {
    d_empty[m] = 1.f;
    for (int n = 0; n < ns; ++n) {
      int p = m * ns + n;
      gidx[p] = 0;
      gx[p * 4 + 0] = 0.f; gx[p * 4 + 1] = 0.f; gx[p * 4 + 2] = 0.f; gx[p * 4 + 3] = 0.f;
    }
  } else {
    d_empty[m] = 0.f;
    float fx = xyz[first * 3 + 0] - qx;
    float fy = xyz[first * 3 + 1] - qy;
    float fz = xyz[first * 3 + 2] - qz;
    for (int n = cnt; n < ns; ++n) {
      int p = m * ns + n;
      gidx[p] = first;
      gx[p * 4 + 0] = fx; gx[p * 4 + 1] = fy; gx[p * 4 + 2] = fz; gx[p * 4 + 3] = 0.f;
    }
  }
}

// ---------------------------------------------------------------------------
// GEMM1: one wave per (16-row, 16-col) tile. Gathers g_feat on the fly,
// computes key_e = relu(X @ Wk^T), v1 = X @ Wv1^T (+ BN1 stats via atomics),
// pos_e = relu(g_xyz @ Wpos^T) in the epilogue (K=3, VALU).
// ---------------------------------------------------------------------------
__global__ void gemm1_kernel(const float* __restrict__ feats,
                             const float* __restrict__ gx,
                             const int* __restrict__ gidx,
                             const float* __restrict__ d_empty,
                             const _Float16* __restrict__ Wk,
                             const _Float16* __restrict__ Wv1,
                             const float* __restrict__ Wpos,
                             _Float16* key16, _Float16* v1_16, _Float16* pos16,
                             float* bsum, float* bsq, int log2ns) {
  int lane = threadIdx.x & 31;
  int row  = blockIdx.x * 16 + (lane & 15);
  int c    = blockIdx.y * 16 + (lane & 15);
  int bk0  = (lane >> 4) << 4;
  float msk = 1.f - d_empty[row >> log2ns];
  int pid = gidx[row];
  // Prefetch the gathered feature row (256B) before the 64 scalar gathers.
  __builtin_prefetch(&feats[pid * 64], 0, 3);
  __builtin_prefetch(&feats[pid * 64 + 32], 0, 3);

  v16h A[3];
#pragma unroll
  for (int ks = 0; ks < 3; ++ks) {
#pragma unroll
    for (int h = 0; h < 16; ++h) {
      int k = ks * 32 + a_k(lane, h);
      float x = 0.f;
      if (k < 3)       x = gx[row * 4 + k];
      else if (k < 67) x = feats[pid * 64 + (k - 3)] * msk;
      A[ks][h] = (_Float16)x;
    }
  }
  v8f ck = {}, cv = {};
#pragma unroll
  for (int ks = 0; ks < 3; ++ks) {
    v16h bk_, bv_;
#pragma unroll
    for (int h = 0; h < 16; ++h) {
      int k = ks * 32 + bk0 + h;
      bk_[h] = Wk[c * 96 + k];
      bv_[h] = Wv1[c * 96 + k];
    }
    ck = WMMA_F16(A[ks], bk_, ck);
    cv = WMMA_F16(A[ks], bv_, cv);
  }
  float p0 = Wpos[c * 3 + 0], p1 = Wpos[c * 3 + 1], p2 = Wpos[c * 3 + 2];
  float s = 0.f, s2 = 0.f;
#pragma unroll
  for (int v = 0; v < 8; ++v) {
    int r = blockIdx.x * 16 + v + ((lane >> 4) << 3);
    key16[r * 128 + c] = (_Float16)fmaxf(ck[v], 0.f);
    float vv = cv[v];
    v1_16[r * 128 + c] = (_Float16)vv;
    s += vv; s2 += vv * vv;
    float pe = p0 * gx[r * 4 + 0] + p1 * gx[r * 4 + 1] + p2 * gx[r * 4 + 2];
    pos16[r * 128 + c] = (_Float16)fmaxf(pe, 0.f);
  }
  s  += __shfl_xor(s, 16);
  s2 += __shfl_xor(s2, 16);
  if (lane < 16) { atomicAdd(&bsum[c], s); atomicAdd(&bsq[c], s2); }
}

// ---------------------------------------------------------------------------
// GEMM2: v2 = relu(BN1(v1)) @ Wv2^T, BN applied on-the-fly to the A operand.
// ---------------------------------------------------------------------------
__global__ void gemm2_kernel(const _Float16* __restrict__ v1,
                             const float* __restrict__ bn1ss,
                             const _Float16* __restrict__ W,
                             _Float16* v2_16, float* bsum, float* bsq) {
  int lane = threadIdx.x & 31;
  int row  = blockIdx.x * 16 + (lane & 15);
  int c    = blockIdx.y * 16 + (lane & 15);
  int bk0  = (lane >> 4) << 4;
  __builtin_prefetch(&v1[row * 128], 0, 3);
  __builtin_prefetch(&v1[row * 128 + 64], 0, 3);
  v8f d = {};
  for (int ks = 0; ks < 128; ks += 32) {
    v16h a, b;
#pragma unroll
    for (int h = 0; h < 16; ++h) {
      int ak = ks + a_k(lane, h);
      float x = (float)v1[row * 128 + ak];
      a[h] = (_Float16)fmaxf(x * bn1ss[ak] + bn1ss[128 + ak], 0.f);
      b[h] = W[c * 128 + ks + bk0 + h];
    }
    d = WMMA_F16(a, b, d);
  }
  float s = 0.f, s2 = 0.f;
#pragma unroll
  for (int v = 0; v < 8; ++v) {
    int r = blockIdx.x * 16 + v + ((lane >> 4) << 3);
    float vv = d[v];
    v2_16[r * 128 + c] = (_Float16)vv;
    s += vv; s2 += vv * vv;
  }
  s  += __shfl_xor(s, 16);
  s2 += __shfl_xor(s2, 16);
  if (lane < 16) { atomicAdd(&bsum[c], s); atomicAdd(&bsq[c], s2); }
}

// Per-query sample means of pos_e, key_e and pos_e*key_e.
__global__ void means_kernel(const _Float16* __restrict__ pos16,
                             const _Float16* __restrict__ key16,
                             _Float16* mP, _Float16* mK, _Float16* mPK, int ns) {
  int m = blockIdx.x, c = threadIdx.x;
  float sp = 0.f, sk = 0.f, spk = 0.f;
  for (int n = 0; n < ns; ++n) {
    int p = (m * ns + n) * 128 + c;
    float pe = (float)pos16[p], ke = (float)key16[p];
    sp += pe; sk += ke; spk += pe * ke;
  }
  float inv = 1.f / (float)ns;
  mP[m * 128 + c]  = (_Float16)(sp * inv);
  mK[m * 128 + c]  = (_Float16)(sk * inv);
  mPK[m * 128 + c] = (_Float16)(spk * inv);
}

// 4 coefficient GEMMs (q/k/qk/v), sigmoid epilogue. blockIdx.z selects which.
__global__ void coef_kernel(const _Float16* mP, const _Float16* mK, const _Float16* mPK,
                            const _Float16* W4, float* coef, int M) {
  int which = blockIdx.z;
  const _Float16* A = (which == 1) ? mK : (which == 2) ? mPK : mP;
  v8f d = wmma_smem_w(A, blockIdx.x * 16, W4 + which * 128 * 128, blockIdx.y * 16, 128);
  int lane = threadIdx.x & 31;
  int c = blockIdx.y * 16 + (lane & 15);
#pragma unroll
  for (int v = 0; v < 8; ++v) {
    int m = blockIdx.x * 16 + v + ((lane >> 4) << 3);
    coef[(which * M + m) * 128 + c] = 1.f / (1.f + __expf(-d[v]));
  }
}

// ---------------------------------------------------------------------------
// Per-query local attention: attn_in / val_e in LDS, 8-head softmax over
// samples, masked aggregation.
// ---------------------------------------------------------------------------
__global__ __launch_bounds__(256) void attn_kernel(
    const _Float16* __restrict__ pos16, const _Float16* __restrict__ key16,
    const _Float16* __restrict__ v2_16, const float* __restrict__ bn2ss,
    const float* __restrict__ coef, const float* __restrict__ d_empty,
    const float* __restrict__ attn_w, float* attend, int ns, int M) {
  __shared__ float s_in[128][33];
  __shared__ float s_val[128][33];
  __shared__ float s_map[8][33];
  __shared__ float s_red[16];
  int m = blockIdx.x, tid = threadIdx.x;
  if (tid < 128) {
    int c = tid;
    float qc  = coef[(0 * M + m) * 128 + c];
    float kc  = coef[(1 * M + m) * 128 + c];
    float qkc = coef[(2 * M + m) * 128 + c];
    float vc  = coef[(3 * M + m) * 128 + c];
    float sc = bn2ss[c], sh = bn2ss[128 + c];
    for (int n = 0; n < ns; ++n) {
      int p = (m * ns + n) * 128 + c;
      float pe = (float)pos16[p];
      float ke = (float)key16[p];
      float ve = fmaxf((float)v2_16[p] * sc + sh, 0.f);
      s_in[c][n]  = pe * qc + ke * kc + pe * ke * qkc;
      s_val[c][n] = ve + pe * vc;
    }
  }
  __syncthreads();
  if (tid < 8 * ns) {
    int h = tid & 7, n = tid >> 3;
    float a = 0.f;
    for (int c = 0; c < 128; ++c) a += attn_w[h * 128 + c] * s_in[c][n];
    s_map[h][n] = a;
  }
  __syncthreads();
  if (tid < 8) {
    float mx = -1e30f;
    for (int n = 0; n < ns; ++n) mx = fmaxf(mx, s_map[tid][n]);
    float s = 0.f;
    for (int n = 0; n < ns; ++n) s += __expf(s_map[tid][n] - mx);
    s_red[tid] = mx; s_red[8 + tid] = 1.f / s;
  }
  __syncthreads();
  if (tid < 8 * ns) {
    int h = tid & 7, n = tid >> 3;
    s_map[h][n] = __expf(s_map[h][n] - s_red[h]) * s_red[8 + h];
  }
  __syncthreads();
  if (tid < 128) {
    int c = tid, h = c >> 4;
    float a = 0.f;
    for (int n = 0; n < ns; ++n) a += s_map[h][n] * s_val[c][n];
    attend[m * 128 + c] = a * (1.f - d_empty[m]);
  }
}

// ---------------------------------------------------------------------------
// LayerNorm over 128 channels, one thread per row.
// ---------------------------------------------------------------------------
__device__ __forceinline__ void ln_rows(float* sF, _Float16* sOut16,
                                        const float* g, const float* b, int G) {
  for (int r = threadIdx.x; r < G; r += 256) {
    float* row = sF + r * 128;
    float mu = 0.f;
    for (int c = 0; c < 128; ++c) mu += row[c];
    mu *= (1.f / 128.f);
    float var = 0.f;
    for (int c = 0; c < 128; ++c) { float d = row[c] - mu; var += d * d; }
    float inv = rsqrtf(var * (1.f / 128.f) + 1e-5f);
    for (int c = 0; c < 128; ++c) {
      float y = (row[c] - mu) * inv * g[c] + b[c];
      row[c] = y;
      if (sOut16) sOut16[r * 128 + c] = (_Float16)y;
    }
  }
}

// ---------------------------------------------------------------------------
// Per-ROI transformer encoder layer + cls/reg heads, all GEMMs on WMMA,
// everything staged in dynamic LDS (<= ~180 KB of the 320 KB WGP pool).
// ---------------------------------------------------------------------------
__global__ __launch_bounds__(256) void transformer_kernel(
    const float* __restrict__ attend, const float* __restrict__ pemb,
    const _Float16* __restrict__ win16, const float* __restrict__ in_b,
    const _Float16* __restrict__ wout16, const float* __restrict__ out_b,
    const _Float16* __restrict__ wlin1, const float* __restrict__ lin1_b,
    const _Float16* __restrict__ wlin2, const float* __restrict__ lin2_b,
    const float* __restrict__ ln1g, const float* __restrict__ ln1b,
    const float* __restrict__ ln2g, const float* __restrict__ ln2b,
    const float* __restrict__ cw1, const float* __restrict__ cb1,
    const float* __restrict__ cw2, const float* __restrict__ cb2,
    const float* __restrict__ rw1, const float* __restrict__ rb1,
    const float* __restrict__ rw2, const float* __restrict__ rb2,
    float* out, int G, int Gp, int lvl_off, int total_cols, int reg_base) {
  extern __shared__ char smem_raw[];
  float*    sF   = (float*)smem_raw;                 // [Gp][128] f (residual stream)
  _Float16* sA   = (_Float16*)(sF + Gp * 128);       // [Gp][128] generic A operand
  _Float16* sB   = sA + Gp * 128;                    // [Gp][128] qk = f + pemb
  _Float16* sQ   = sB + Gp * 128;                    // [Gp][128]
  _Float16* sK   = sQ + Gp * 128;                    // [Gp][128]
  _Float16* sV   = sK + Gp * 128;                    // [Gp][128]
  float*    sAtt = (float*)(sV + Gp * 128);          // [4][Gp][Gp]
  float*    sCW  = sAtt + 4 * Gp * Gp;               // [Gp]
  float*    sRW  = sCW + Gp;                         // [Gp]

  const int tid = threadIdx.x, wave = tid >> 5, lane = tid & 31;
  const int roi = blockIdx.x, R = Gp >> 4;

  // stage 0: load f, qk
  for (int i = tid; i < Gp * 128; i += 256) {
    int g = i >> 7, c = i & 127;
    float v = 0.f, pe = 0.f;
    if (g < G) { v = attend[(roi * G + g) * 128 + c]; pe = pemb[g * 128 + c]; }
    sF[i] = v;
    sA[i] = (_Float16)v;
    sB[i] = (_Float16)(v + pe);
  }
  __syncthreads();

  // stage 1: QKV = [qk,qk,f] @ win^T + b ; q scaled by hd^-0.5
  for (int t = wave; t < R * 24; t += 8) {
    int rt = t / 24, ct = t % 24;
    const _Float16* Asrc = (ct >= 16) ? sA : sB;
    v8f d = wmma_smem_w(Asrc, rt * 16, win16, ct * 16, 128);
    int co = ct * 16 + (lane & 15);
    float bias = in_b[co];
#pragma unroll
    for (int v = 0; v < 8; ++v) {
      int r = rt * 16 + v + ((lane >> 4) << 3);
      float val = d[v] + bias;
      if (co < 128)       sQ[r * 128 + co] = (_Float16)(val * 0.17677669529663687f);
      else if (co < 256)  sK[r * 128 + (co - 128)] = (_Float16)val;
      else                sV[r * 128 + (co - 256)] = (_Float16)val;
    }
  }
  __syncthreads();

  // stage 2: logits = Qh @ Kh^T per head (K = hd = 32 -> one WMMA k-step)
  for (int t = wave; t < 4 * R * R; t += 8) {
    int h4 = t / (R * R), rem = t % (R * R), rt = rem / R, ct = rem % R;
    int hoff = h4 * 32;
    int arow = (rt * 16 + (lane & 15)) * 128;
    int brow = (ct * 16 + (lane & 15)) * 128;
    int bk0 = (lane >> 4) << 4;
    v16h a, b;
#pragma unroll
    for (int h = 0; h < 16; ++h) {
      a[h] = sQ[arow + hoff + a_k(lane, h)];
      b[h] = sK[brow + hoff + bk0 + h];
    }
    v8f d = {};
    d = WMMA_F16(a, b, d);
    int gc = ct * 16 + (lane & 15);
#pragma unroll
    for (int v = 0; v < 8; ++v) {
      int gr = rt * 16 + v + ((lane >> 4) << 3);
      sAtt[(h4 * Gp + gr) * Gp + gc] = (gc < G) ? d[v] : -1e30f;
    }
  }
  __syncthreads();

  // stage 3: row softmax per head (padded rows/cols -> 0 prob)
  for (int t = tid; t < 4 * Gp; t += 256) {
    int h4 = t / Gp, r = t % Gp;
    float* row = sAtt + (h4 * Gp + r) * Gp;
    if (r < G) {
      float mx = -1e30f;
      for (int n = 0; n < G; ++n) mx = fmaxf(mx, row[n]);
      float s = 0.f;
      for (int n = 0; n < G; ++n) { float e = __expf(row[n] - mx); row[n] = e; s += e; }
      float inv = 1.f / s;
      for (int n = 0; n < Gp; ++n) row[n] = (n < G) ? row[n] * inv : 0.f;
    } else {
      for (int n = 0; n < Gp; ++n) row[n] = 0.f;
    }
  }
  __syncthreads();

  // stage 4: O = P @ V per head -> sA (f16)
  for (int t = wave; t < 4 * R * 2; t += 8) {
    int h4 = t / (R * 2), rem = t % (R * 2), rt = rem / 2, ct2 = rem % 2;
    int cch = h4 * 32 + ct2 * 16 + (lane & 15);
    int arow = rt * 16 + (lane & 15);
    int bk0 = (lane >> 4) << 4;
    v8f d = {};
    for (int ks = 0; ks < Gp; ks += 32) {
      v16h a, b;
#pragma unroll
      for (int h = 0; h < 16; ++h) {
        a[h] = (_Float16)sAtt[(h4 * Gp + arow) * Gp + ks + a_k(lane, h)];
        b[h] = sV[(ks + bk0 + h) * 128 + cch];
      }
      d = WMMA_F16(a, b, d);
    }
#pragma unroll
    for (int v = 0; v < 8; ++v) {
      int r = rt * 16 + v + ((lane >> 4) << 3);
      sA[r * 128 + cch] = (_Float16)d[v];
    }
  }
  __syncthreads();

  // stage 5: out-projection + residual into sF
  for (int t = wave; t < R * 8; t += 8) {
    int rt = t / 8, ct = t % 8;
    v8f d = wmma_smem_w(sA, rt * 16, wout16, ct * 16, 128);
    int co = ct * 16 + (lane & 15);
    float bias = out_b[co];
#pragma unroll
    for (int v = 0; v < 8; ++v) {
      int r = rt * 16 + v + ((lane >> 4) << 3);
      sF[r * 128 + co] += d[v] + bias;
    }
  }
  __syncthreads();

  // stage 6: LN1 -> sF (f32) + sA (f16)
  ln_rows(sF, sA, ln1g, ln1b, G);
  __syncthreads();

  // stage 7: FF1 = relu(f @ w1^T + b1) -> sQ
  for (int t = wave; t < R * 8; t += 8) {
    int rt = t / 8, ct = t % 8;
    v8f d = wmma_smem_w(sA, rt * 16, wlin1, ct * 16, 128);
    int co = ct * 16 + (lane & 15);
    float bias = lin1_b[co];
#pragma unroll
    for (int v = 0; v < 8; ++v) {
      int r = rt * 16 + v + ((lane >> 4) << 3);
      sQ[r * 128 + co] = (_Float16)fmaxf(d[v] + bias, 0.f);
    }
  }
  __syncthreads();

  // stage 8: FF2 + residual
  for (int t = wave; t < R * 8; t += 8) {
    int rt = t / 8, ct = t % 8;
    v8f d = wmma_smem_w(sQ, rt * 16, wlin2, ct * 16, 128);
    int co = ct * 16 + (lane & 15);
    float bias = lin2_b[co];
#pragma unroll
    for (int v = 0; v < 8; ++v) {
      int r = rt * 16 + v + ((lane >> 4) << 3);
      sF[r * 128 + co] += d[v] + bias;
    }
  }
  __syncthreads();

  // stage 9: LN2 -> final f
  ln_rows(sF, nullptr, ln2g, ln2b, G);
  __syncthreads();

  // stage 10: cls/reg gating heads (small, VALU)
  for (int r = tid; r < G; r += 256) {
    const float* frow = sF + r * 128;
    float sc = cb2[0], sr = rb2[0];
    for (int j = 0; j < 64; ++j) {
      float a1 = cb1[j], a2 = rb1[j];
      for (int c = 0; c < 128; ++c) {
        float f = frow[c];
        a1 += f * cw1[j * 128 + c];
        a2 += f * rw1[j * 128 + c];
      }
      sc += fmaxf(a1, 0.f) * cw2[j];
      sr += fmaxf(a2, 0.f) * rw2[j];
    }
    sCW[r] = 1.f / (1.f + __expf(-sc));
    sRW[r] = 1.f / (1.f + __expf(-sr));
  }
  __syncthreads();

  // stage 11: final outputs
  for (int i = tid; i < G * 128; i += 256) {
    int r = i >> 7, c = i & 127;
    float f = sF[r * 128 + c];
    int col = lvl_off + r * 128 + c;
    out[roi * total_cols + col] = f * sCW[r];
    out[reg_base + roi * total_cols + col] = f * sRW[r];
  }
}

// ---------------------------------------------------------------------------
// Host orchestration
// ---------------------------------------------------------------------------
extern "C" void kernel_launch(void* const* d_in, const int* in_sizes, int n_in,
                              void* d_out, int out_size, void* d_ws, size_t ws_size,
                              hipStream_t stream) {
  (void)in_sizes; (void)n_in; (void)out_size; (void)ws_size;
  const float* xyz      = (const float*)d_in[0];
  const float* feats    = (const float*)d_in[1];
  const int*   xyz_cnt  = (const int*)d_in[2];
  const float* q_xyz[2] = {(const float*)d_in[3], (const float*)d_in[7]};
  const float* q_r[2]   = {(const float*)d_in[4], (const float*)d_in[8]};
  const int*   q_cnt[2] = {(const int*)d_in[5], (const int*)d_in[9]};
  const float* pemb[2]  = {(const float*)d_in[6], (const float*)d_in[10]};
  const float* pos_w    = (const float*)d_in[11];
  const float* key_w    = (const float*)d_in[12];
  const float* vw1      = (const float*)d_in[13];
  const float* bn1g     = (const float*)d_in[14];
  const float* bn1b     = (const float*)d_in[15];
  const float* vw2      = (const float*)d_in[16];
  const float* bn2g     = (const float*)d_in[17];
  const float* bn2b     = (const float*)d_in[18];
  const float* attn_w   = (const float*)d_in[19];
  const float* coef_w[4] = {(const float*)d_in[20], (const float*)d_in[21],
                            (const float*)d_in[22], (const float*)d_in[23]};
  const float* in_w  = (const float*)d_in[24];
  const float* in_b  = (const float*)d_in[25];
  const float* out_w = (const float*)d_in[26];
  const float* out_b = (const float*)d_in[27];
  const float* l1w   = (const float*)d_in[28];
  const float* l1b   = (const float*)d_in[29];
  const float* l2w   = (const float*)d_in[30];
  const float* l2b   = (const float*)d_in[31];
  const float* ln1g  = (const float*)d_in[32];
  const float* ln1b  = (const float*)d_in[33];
  const float* ln2g  = (const float*)d_in[34];
  const float* ln2b  = (const float*)d_in[35];
  const float* cw1 = (const float*)d_in[36];
  const float* cb1 = (const float*)d_in[37];
  const float* cw2 = (const float*)d_in[38];
  const float* cb2 = (const float*)d_in[39];
  const float* rw1 = (const float*)d_in[40];
  const float* rb1 = (const float*)d_in[41];
  const float* rw2 = (const float*)d_in[42];
  const float* rb2 = (const float*)d_in[43];
  float* out = (float*)d_out;

  const int NPTS = 8192, MAXM = 8192, MAXP = 262144;
  const int TOTAL_COLS = 27 * 128 + 64 * 128;  // 11648
  const int REG_BASE = 128 * TOTAL_COLS;

  // workspace bump allocator
  char* ws = (char*)d_ws;
  size_t off = 0;
  auto alloc = [&](size_t bytes) -> void* {
    off = (off + 255) & ~(size_t)255;
    void* p = ws + off;
    off += bytes;
    return p;
  };
  _Float16* wk16    = (_Float16*)alloc(2 * 128 * 96 * 2);
  _Float16* wv1_16  = (_Float16*)alloc(2 * 128 * 96 * 2);
  _Float16* wv2_16  = (_Float16*)alloc(2 * 128 * 128 * 2);
  _Float16* wcoef16 = (_Float16*)alloc(2 * 4 * 128 * 128 * 2);
  _Float16* win16   = (_Float16*)alloc(2 * 384 * 128 * 2);
  _Float16* wout16  = (_Float16*)alloc(2 * 128 * 128 * 2);
  _Float16* wl1_16  = (_Float16*)alloc(2 * 128 * 128 * 2);
  _Float16* wl2_16  = (_Float16*)alloc(2 * 128 * 128 * 2);
  float*    gx      = (float*)alloc((size_t)MAXP * 4 * 4);
  int*      gidx    = (int*)alloc((size_t)MAXP * 4);
  float*    d_empty = (float*)alloc((size_t)MAXM * 4);
  _Float16* pos16   = (_Float16*)alloc((size_t)MAXP * 128 * 2);
  _Float16* key16   = (_Float16*)alloc((size_t)MAXP * 128 * 2);
  _Float16* v1_16   = (_Float16*)alloc((size_t)MAXP * 128 * 2);
  _Float16* v2_16   = (_Float16*)alloc((size_t)MAXP * 128 * 2);
  _Float16* meanP   = (_Float16*)alloc((size_t)MAXM * 128 * 2);
  _Float16* meanK   = (_Float16*)alloc((size_t)MAXM * 128 * 2);
  _Float16* meanPK  = (_Float16*)alloc((size_t)MAXM * 128 * 2);
  float*    coefb   = (float*)alloc((size_t)4 * MAXM * 128 * 4);
  float*    attend  = (float*)alloc((size_t)MAXM * 128 * 4);
  float*    bnacc   = (float*)alloc(512 * 4);  // bn1 sum/sq, bn2 sum/sq
  float*    bn1ss   = (float*)alloc(256 * 4);
  float*    bn2ss   = (float*)alloc(256 * 4);

  auto cvt = [&](const float* src, _Float16* dst, int rows, int sk, int dk) {
    int n = rows * dk;
    cvt_kernel<<<(n + 255) / 256, 256, 0, stream>>>(src, dst, rows, sk, dk);
  };
  for (int lvl = 0; lvl < 2; ++lvl) {
    cvt(key_w + lvl * 128 * 67, wk16 + lvl * 128 * 96, 128, 67, 96);
    cvt(vw1 + lvl * 128 * 67, wv1_16 + lvl * 128 * 96, 128, 67, 96);
    cvt(vw2 + lvl * 16384, wv2_16 + lvl * 16384, 128, 128, 128);
    for (int w = 0; w < 4; ++w)
      cvt(coef_w[w] + lvl * 16384, wcoef16 + (lvl * 4 + w) * 16384, 128, 128, 128);
    cvt(in_w + lvl * 49152, win16 + lvl * 49152, 384, 128, 128);
    cvt(out_w + lvl * 16384, wout16 + lvl * 16384, 128, 128, 128);
    cvt(l1w + lvl * 16384, wl1_16 + lvl * 16384, 128, 128, 128);
    cvt(l2w + lvl * 16384, wl2_16 + lvl * 16384, 128, 128, 128);
  }

  int lvl_off = 0;
  for (int lvl = 0; lvl < 2; ++lvl) {
    const int gs = (lvl == 0) ? 3 : 4;
    const int G = gs * gs * gs;
    const int Gp = (G + 15) & ~15;
    const int ns = (lvl == 0) ? 16 : 32;
    const int log2ns = (lvl == 0) ? 4 : 5;
    const int M = 2 * 64 * G;
    const int P = M * ns;

    group_kernel<<<(M + 255) / 256, 256, 0, stream>>>(
        xyz, q_xyz[lvl], q_r[lvl], xyz_cnt, q_cnt[lvl], gx, gidx, d_empty, M, ns, NPTS);

    zero_kernel<<<2, 256, 0, stream>>>(bnacc, 512);

    gemm1_kernel<<<dim3(P / 16, 8), 32, 0, stream>>>(
        feats, gx, gidx, d_empty,
        wk16 + lvl * 128 * 96, wv1_16 + lvl * 128 * 96, pos_w + lvl * 128 * 3,
        key16, v1_16, pos16, bnacc, bnacc + 128, log2ns);

    bn_final_kernel<<<1, 128, 0, stream>>>(bnacc, bnacc + 128,
                                           bn1g + lvl * 128, bn1b + lvl * 128,
                                           bn1ss, 1.f / (float)P);

    gemm2_kernel<<<dim3(P / 16, 8), 32, 0, stream>>>(
        v1_16, bn1ss, wv2_16 + lvl * 16384, v2_16, bnacc + 256, bnacc + 384);

    bn_final_kernel<<<1, 128, 0, stream>>>(bnacc + 256, bnacc + 384,
                                           bn2g + lvl * 128, bn2b + lvl * 128,
                                           bn2ss, 1.f / (float)P);

    means_kernel<<<M, 128, 0, stream>>>(pos16, key16, meanP, meanK, meanPK, ns);

    coef_kernel<<<dim3(M / 16, 8, 4), 32, 0, stream>>>(
        meanP, meanK, meanPK, wcoef16 + lvl * 4 * 16384, coefb, M);

    attn_kernel<<<M, 256, 0, stream>>>(pos16, key16, v2_16, bn2ss, coefb, d_empty,
                                       attn_w + lvl * 8 * 128, attend, ns, M);

    size_t smem = (size_t)Gp * 128 * 4 + (size_t)Gp * 128 * 2 * 5 +
                  (size_t)4 * Gp * Gp * 4 + (size_t)2 * Gp * 4;
    transformer_kernel<<<128, 256, smem, stream>>>(
        attend, pemb[lvl],
        win16 + lvl * 49152, in_b + lvl * 384,
        wout16 + lvl * 16384, out_b + lvl * 128,
        wl1_16 + lvl * 16384, l1b + lvl * 128,
        wl2_16 + lvl * 16384, l2b + lvl * 128,
        ln1g + lvl * 128, ln1b + lvl * 128,
        ln2g + lvl * 128, ln2b + lvl * 128,
        cw1 + lvl * 64 * 128, cb1 + lvl * 64, cw2 + lvl * 64, cb2 + lvl,
        rw1 + lvl * 64 * 128, rb1 + lvl * 64, rw2 + lvl * 64, rb2 + lvl,
        out, G, Gp, lvl_off, TOTAL_COLS, REG_BASE);

    lvl_off += G * 128;
  }
}
